// PointNetSetAbstractionMsg_29394756174197
// MI455X (gfx1250) — matrix operand order repaired and checked by reading, not compiled, not run on hardware
//
#include <hip/hip_runtime.h>

// ---------------- problem constants ----------------
#define B_    16
#define N_    2048
#define S_    512
#define INCH  64
#define C0    67      // 3 (rel xyz) + 64 (features)
#define C0PAD 96      // padded to multiple of 32 for WMMA K-steps

typedef _Float16 half_t;
typedef __attribute__((ext_vector_type(16))) _Float16 v16h;
typedef __attribute__((ext_vector_type(8)))  _Float16 v8h;
typedef __attribute__((ext_vector_type(8)))  float    v8f;

// =====================================================================
// 1) Farthest point sampling: one block per batch, whole cloud in LDS.
// =====================================================================
__global__ void fps_kernel(const float* __restrict__ xyz,   // [B][3][N]
                           int*   __restrict__ fps_idx,     // [B][S]
                           float* __restrict__ new_xyz)     // [B][S][3]
{
    __shared__ float sx[N_], sy[N_], sz[N_], dist[N_];
    __shared__ float rv[8];
    __shared__ int   ri[8];
    __shared__ int   s_far;
    const int b = blockIdx.x, tid = threadIdx.x;
    const float* base = xyz + (long)b * 3 * N_;
    for (int i = tid; i < N_; i += 256) {
        sx[i] = base[i]; sy[i] = base[N_ + i]; sz[i] = base[2 * N_ + i];
        dist[i] = 1e10f;
    }
    if (tid == 0) s_far = 0;
    __syncthreads();
    for (int it = 0; it < S_; ++it) {
        const int far = s_far;
        const float fx = sx[far], fy = sy[far], fz = sz[far];
        if (tid == 0) {
            fps_idx[b * S_ + it] = far;
            new_xyz[((long)b * S_ + it) * 3 + 0] = fx;
            new_xyz[((long)b * S_ + it) * 3 + 1] = fy;
            new_xyz[((long)b * S_ + it) * 3 + 2] = fz;
        }
        float best = -1.0f; int bi = 0;
        for (int i = tid; i < N_; i += 256) {
            const float dx = sx[i] - fx, dy = sy[i] - fy, dz = sz[i] - fz;
            const float d = dx * dx + dy * dy + dz * dz;
            const float m = dist[i] < d ? dist[i] : d;
            dist[i] = m;
            if (m > best) { best = m; bi = i; }
        }
        // wave32 argmax reduce
        for (int off = 16; off > 0; off >>= 1) {
            const float ob = __shfl_down(best, off, 32);
            const int   oi = __shfl_down(bi,   off, 32);
            if (ob > best || (ob == best && oi < bi)) { best = ob; bi = oi; }
        }
        if ((tid & 31) == 0) { rv[tid >> 5] = best; ri[tid >> 5] = bi; }
        __syncthreads();
        if (tid == 0) {
            float bb = rv[0]; int bbi = ri[0];
            for (int w = 1; w < 8; ++w)
                if (rv[w] > bb || (rv[w] == bb && ri[w] < bbi)) { bb = rv[w]; bbi = ri[w]; }
            s_far = bbi;
        }
        __syncthreads();
    }
}

// =====================================================================
// 2) Radius NN: one wave per query; ballot-compaction keeps the K
//    lowest-index in-radius points (matches the reference's sort trick).
// =====================================================================
__global__ void radius_nn_kernel(const float* __restrict__ xyz,      // [B][3][N]
                                 const float* __restrict__ new_xyz,  // [B][S][3]
                                 int* __restrict__ nn,               // [B*S][K]
                                 float r2, int K)
{
    const int lane = threadIdx.x & 31;
    const int q    = blockIdx.x * 8 + (threadIdx.x >> 5);   // b*S+s
    const int b    = q / S_;
    const float* xb = xyz + (long)b * 3 * N_;
    const float qx = new_xyz[q * 3 + 0];
    const float qy = new_xyz[q * 3 + 1];
    const float qz = new_xyz[q * 3 + 2];
    int* out = nn + (long)q * K;
    int cnt = 0, firstIdx = 0;
    for (int i0 = 0; i0 < N_ && cnt < K; i0 += 32) {
        const int i = i0 + lane;
        const float dx = xb[i] - qx, dy = xb[N_ + i] - qy, dz = xb[2 * N_ + i] - qz;
        const float d = dx * dx + dy * dy + dz * dz;
        const unsigned m = __builtin_amdgcn_ballot_w32(d <= r2);
        const int pre = __popc(m & ((1u << lane) - 1u));
        if ((m >> lane) & 1u) {
            const int pos = cnt + pre;
            if (pos < K) out[pos] = i;
        }
        if (cnt == 0 && m) firstIdx = i0 + __ffs((int)m) - 1;
        cnt += __popc(m);
    }
    if (cnt > K) cnt = K;
    for (int j = cnt + lane; j < K; j += 32) out[j] = firstIdx;  // pad with first
}

// =====================================================================
// 3) Group: build fp16 X0 [C0PAD][P], P = B*S*K.  rows 0..2 rel-xyz,
//    3..66 gathered features, 67..95 zero padding.
// =====================================================================
__global__ void group_kernel(const float* __restrict__ xyz,
                             const float* __restrict__ feat,
                             const float* __restrict__ new_xyz,
                             const int*   __restrict__ nn,
                             half_t* __restrict__ X0, int K, long P)
{
    const long p = (long)blockIdx.x * 256 + threadIdx.x;
    if (p >= P) return;
    const long q  = p / K;                    // b*S+s
    const int  b  = (int)(q / S_);
    const int idx = nn[p];
    const float* xb = xyz  + (long)b * 3 * N_;
    const float* fb = feat + (long)b * INCH * N_;
#pragma unroll
    for (int c = 0; c < 3; ++c)
        X0[(long)c * P + p] = (half_t)(xb[c * N_ + idx] - new_xyz[q * 3 + c]);
    for (int c = 0; c < INCH; ++c)
        X0[(long)(3 + c) * P + p] = (half_t)fb[c * N_ + idx];
    for (int c = C0; c < C0PAD; ++c)
        X0[(long)c * P + p] = (half_t)0.0f;
}

// =====================================================================
// 4) WMMA GEMM + fused input-BN/ReLU + per-channel sum/sumsq epilogue.
//    Y[o,p] = W[o,:] . bnrelu(X[:,p]) + bias[o]
//    Block = 8 waves, 128-column tile; each wave owns 16 columns.
//    Templated on NK = Cin/32 so the wave's full B operand (NK v16h
//    fragments) lives in registers across the whole row-tile loop:
//    inner loop is 2x ds_load_b128 (A) + 1x v_wmma_f32_16x16x32_f16.
// =====================================================================
template <int NK>
__global__ void gemm_bn_kernel(const half_t* __restrict__ Xin,   // [Cin][P] fp16
                               const float*  __restrict__ W,     // [Cout][CinReal] f32
                               const float*  __restrict__ bias,  // [Cout]
                               const float*  __restrict__ inScale, // [Cin] (BN of prev layer)
                               const float*  __restrict__ inShift,
                               int applyBn,
                               half_t* __restrict__ Yout,        // [Cout][P] fp16
                               float* __restrict__ gSum, float* __restrict__ gSq,
                               int CinReal, int Cout, long P)
{
    constexpr int Cin = NK * 32;
    __shared__ half_t Wl[128 * 96];   // weights, row-major [o][Cin], fp16, zero-padded
    __shared__ half_t Xt[128 * 96];   // X tile, transposed: [col][Cin]
    __shared__ float  sSum[128], sSq[128];
    const int tid = threadIdx.x;
    const long p0 = (long)blockIdx.x * 128;

    if (tid < 128) { sSum[tid] = 0.0f; sSq[tid] = 0.0f; }
    for (int i = tid; i < Cout * Cin; i += 256) {
        const int o = i / Cin, c = i % Cin;
        Wl[i] = (half_t)((c < CinReal) ? W[(long)o * CinReal + c] : 0.0f);
    }
    for (int i = tid; i < 128 * Cin; i += 256) {
        const int c = i >> 7, col = i & 127;
        float xv = (float)Xin[(long)c * P + p0 + col];
        if (applyBn) xv = fmaxf(xv * inScale[c] + inShift[c], 0.0f);
        Xt[col * Cin + c] = (half_t)xv;
        if (p0 + 128 < P)   // speculative prefetch of next column tile -> global_prefetch_b8
            __builtin_prefetch(Xin + (long)c * P + p0 + 128 + col, 0, 1);
    }
    __syncthreads();

    const int lane     = tid & 31;
    const int wid      = tid >> 5;                   // 0..7
    const int colLocal = wid * 16 + (lane & 15);
    const int selA     = (lane < 16) ? 0 : 8;        // A frag: lanes 16-31 hold K=8..15 / 24..31
    const int selB     = (lane < 16) ? 0 : 8;        // B frag: lanes 16-31 hold K=8..15 / 24..31
    const half_t* xbase = &Xt[colLocal * Cin];

    // hoist the wave's whole B operand into registers (reused by every row tile)
    v16h bf[NK];
#pragma unroll
    for (int k = 0; k < NK; ++k) {
        const v8h blo = *(const v8h*)(xbase + k * 32 + selB);
        const v8h bhi = *(const v8h*)(xbase + k * 32 + selB + 16);
        bf[k] = __builtin_shufflevector(blo, bhi,
                    0,1,2,3,4,5,6,7,8,9,10,11,12,13,14,15);
    }

    for (int mt = 0; mt < (Cout >> 4); ++mt) {
        v8f acc = {};
        const int wrow = mt * 16 + (lane & 15);
        const half_t* wbase = &Wl[wrow * Cin];
#pragma unroll
        for (int k = 0; k < NK; ++k) {
            const v8h alo = *(const v8h*)(wbase + k * 32 + selA);
            const v8h ahi = *(const v8h*)(wbase + k * 32 + selA + 16);
            const v16h a  = __builtin_shufflevector(alo, ahi,
                              0,1,2,3,4,5,6,7,8,9,10,11,12,13,14,15);
            acc = __builtin_amdgcn_wmma_f32_16x16x32_f16(
                      false, a, false, bf[k], (short)0, acc, false, false);
        }
        const long pg   = p0 + colLocal;
        const int rbase = mt * 16 + ((lane < 16) ? 0 : 8);
#pragma unroll
        for (int r = 0; r < 8; ++r) {
            const int row = rbase + r;
            const float y = acc[r] + bias[row];
            Yout[(long)row * P + pg] = (half_t)y;
            atomicAdd(&sSum[row], y);
            atomicAdd(&sSq[row],  y * y);
        }
    }
    __syncthreads();
    if (tid < Cout) {
        atomicAdd(&gSum[tid], sSum[tid]);
        atomicAdd(&gSq[tid],  sSq[tid]);
    }
}

// =====================================================================
// 5) BN stats finalize -> per-channel affine (scale, shift)
// =====================================================================
__global__ void bn_finalize_kernel(const float* gSum, const float* gSq,
                                   const float* gamma, const float* beta,
                                   float* scalev, float* shiftv,
                                   int Cout, float invCount)
{
    const int o = threadIdx.x;
    if (o < Cout) {
        const float mu  = gSum[o] * invCount;
        const float var = gSq[o] * invCount - mu * mu;
        const float sc  = gamma[o] * rsqrtf(var + 1e-5f);
        scalev[o] = sc;
        shiftv[o] = beta[o] - mu * sc;
    }
}

// =====================================================================
// 6) Final: BN + ReLU + max over K -> output feature slab
// =====================================================================
__global__ void final_max_kernel(const half_t* __restrict__ Y,   // [Cout][P]
                                 const float* scalev, const float* shiftv,
                                 float* __restrict__ outFeat,    // [B][320][S]
                                 int Cout, int chOff, int K, long P)
{
    const long t = (long)blockIdx.x * 256 + threadIdx.x;
    const long total = (long)B_ * S_ * Cout;
    if (t >= total) return;
    const int s = (int)(t % S_);
    const long r = t / S_;
    const int o = (int)(r % Cout);
    const int b = (int)(r / Cout);
    const half_t* yp = Y + (long)o * P + ((long)(b * S_ + s)) * K;
    const float sc = scalev[o], sh = shiftv[o];
    float m = -1e30f;
    for (int k = 0; k < K; ++k) m = fmaxf(m, (float)yp[k] * sc + sh);
    m = fmaxf(m, 0.0f);   // relu then max == max then relu (monotone)
    outFeat[((long)b * 320 + (chOff + o)) * S_ + s] = m;
}

// =====================================================================
// 7) out_xyz writer: (B,S,3) -> (B,3,S)
// =====================================================================
__global__ void write_xyz_kernel(const float* __restrict__ new_xyz, float* __restrict__ out)
{
    const int t = blockIdx.x * 256 + threadIdx.x;   // < B*3*S
    const int s = t % S_;
    const int r = t / S_;
    const int d = r % 3;
    const int b = r / 3;
    out[(b * 3 + d) * S_ + s] = new_xyz[((long)b * S_ + s) * 3 + d];
}

__global__ void zero_kernel(float* p, int n)
{
    const int t = blockIdx.x * 256 + threadIdx.x;
    if (t < n) p[t] = 0.0f;
}

// =====================================================================
// host launcher
// =====================================================================
extern "C" void kernel_launch(void* const* d_in, const int* in_sizes, int n_in,
                              void* d_out, int out_size, void* d_ws, size_t ws_size,
                              hipStream_t stream)
{
    const float* points_xyz  = (const float*)d_in[0];
    const float* points_feat = (const float*)d_in[1];
    // params: d_in[2 + (i*3+j)*4 + {0:w,1:b,2:g,3:be}]
    auto Wp  = [&](int i, int j) { return (const float*)d_in[2 + (i * 3 + j) * 4 + 0]; };
    auto Bp  = [&](int i, int j) { return (const float*)d_in[2 + (i * 3 + j) * 4 + 1]; };
    auto Gp  = [&](int i, int j) { return (const float*)d_in[2 + (i * 3 + j) * 4 + 2]; };
    auto BEp = [&](int i, int j) { return (const float*)d_in[2 + (i * 3 + j) * 4 + 3]; };

    // workspace layout
    char* ws = (char*)d_ws;
    half_t* bufA = (half_t*)ws;                              // 128 * 1048576 fp16 = 256 MB
    half_t* bufB = (half_t*)(ws + 268435456ll);              // 256 MB
    char* aux = ws + 536870912ll;
    int*   fps_idx = (int*)aux;           aux += 32768;      // 16*512 int
    float* new_xyz = (float*)aux;         aux += 98304;      // 16*512*3 f32
    int*   nn      = (int*)aux;           aux += 4194304;    // 16*512*128 int
    float* gSum    = (float*)aux;         aux += 128 * 4;
    float* gSq     = (float*)aux;         aux += 128 * 4;
    float* sc_     = (float*)aux;         aux += 3 * 128 * 4;
    float* sh_     = (float*)aux;         aux += 3 * 128 * 4;

    const int   Ks[3]      = {16, 32, 128};
    const float r2s[3]     = {0.01f, 0.04f, 0.16f};
    const int   mlp[3][3]  = {{32, 32, 64}, {64, 64, 128}, {64, 96, 128}};
    const int   chOff[3]   = {0, 64, 192};

    fps_kernel<<<B_, 256, 0, stream>>>(points_xyz, fps_idx, new_xyz);
    write_xyz_kernel<<<(B_ * 3 * S_) / 256, 256, 0, stream>>>(new_xyz, (float*)d_out);
    float* outFeat = (float*)d_out + (long)B_ * 3 * S_;

    for (int sidx = 0; sidx < 3; ++sidx) {
        const int  K = Ks[sidx];
        const long P = (long)B_ * S_ * K;
        radius_nn_kernel<<<(B_ * S_) / 8, 256, 0, stream>>>(points_xyz, new_xyz, nn, r2s[sidx], K);
        group_kernel<<<(unsigned)(P / 256), 256, 0, stream>>>(points_xyz, points_feat,
                                                              new_xyz, nn, bufA, K, P);
        half_t* X = bufA;
        half_t* Y = bufB;
        int Cin = C0PAD, CinReal = C0;
        for (int j = 0; j < 3; ++j) {
            const int Cout = mlp[sidx][j];
            zero_kernel<<<1, 256, 0, stream>>>(gSum, 128);
            zero_kernel<<<1, 256, 0, stream>>>(gSq, 128);
            const float* isc = (j > 0) ? (sc_ + (j - 1) * 128) : nullptr;
            const float* ish = (j > 0) ? (sh_ + (j - 1) * 128) : nullptr;
            const int    abn = (j > 0) ? 1 : 0;
            const unsigned nb = (unsigned)(P / 128);
            switch (Cin >> 5) {
            case 1:
                gemm_bn_kernel<1><<<nb, 256, 0, stream>>>(X, Wp(sidx, j), Bp(sidx, j),
                    isc, ish, abn, Y, gSum, gSq, CinReal, Cout, P);
                break;
            case 2:
                gemm_bn_kernel<2><<<nb, 256, 0, stream>>>(X, Wp(sidx, j), Bp(sidx, j),
                    isc, ish, abn, Y, gSum, gSq, CinReal, Cout, P);
                break;
            default:
                gemm_bn_kernel<3><<<nb, 256, 0, stream>>>(X, Wp(sidx, j), Bp(sidx, j),
                    isc, ish, abn, Y, gSum, gSq, CinReal, Cout, P);
                break;
            }
            bn_finalize_kernel<<<1, 128, 0, stream>>>(gSum, gSq, Gp(sidx, j), BEp(sidx, j),
                                                      sc_ + j * 128, sh_ + j * 128,
                                                      Cout, 1.0f / (float)P);
            half_t* t = X; X = Y; Y = t;
            Cin = Cout; CinReal = Cout;
        }
        const int  CoutLast = mlp[sidx][2];
        const long total    = (long)B_ * S_ * CoutLast;
        final_max_kernel<<<(unsigned)((total + 255) / 256), 256, 0, stream>>>(
            X, sc_ + 2 * 128, sh_ + 2 * 128, outFeat, CoutLast, chOff[sidx], K, P);
    }
}
